// MultiHeadAttention_47914655154223
// MI455X (gfx1250) — matrix operand order repaired and checked by reading, not compiled, not run on hardware
//
#include <hip/hip_runtime.h>
#include <hip/hip_bf16.h>
#include <math.h>

// MI455X / gfx1250, wave32. Matrix math via v_wmma_f32_16x16x32_bf16;
// A-tile staging via the Tensor Data Mover (tensor_load_to_lds + TENSORcnt).

typedef __attribute__((ext_vector_type(16))) __bf16 v16bf;
typedef __attribute__((ext_vector_type(8)))  float  v8f;
typedef __attribute__((ext_vector_type(4)))  unsigned int u32x4;
typedef __attribute__((ext_vector_type(8)))  int          i32x8;
typedef __attribute__((ext_vector_type(4)))  int          i32x4;

#define GRAPH  512
#define NBATCH 128
#define DIM    512
#define ROWS   (NBATCH * GRAPH)   // 65536

// ---------------------------------------------------------------- fp32 -> bf16
__global__ void mha_cvt_bf16(const float* __restrict__ src,
                             __bf16* __restrict__ dst, int n) {
  int i = blockIdx.x * blockDim.x + threadIdx.x;
  if (i < n) dst[i] = (__bf16)src[i];
}

// ------------------------------------------------------------- TDM tile load
// 2D tile (rows x cols bf16, row stride = stride_elems) -> LDS, via D#
// descriptor (CDNA5 ISA). All inputs must be uniform.
__device__ __forceinline__ void tdm_load_tile_bf16(
    unsigned lds_addr, const void* gptr, unsigned rows, unsigned cols,
    unsigned stride_elems) {
  const unsigned long long ga = (unsigned long long)(uintptr_t)gptr;
  u32x4 g0;
  g0[0] = 1u;                                    // count=1, user descriptor
  g0[1] = lds_addr;                              // D#.lds_addr (bytes)
  g0[2] = (unsigned)(ga & 0xffffffffu);          // global_addr[31:0]
  g0[3] = (unsigned)((ga >> 32) & 0x01ffffffu)   // global_addr[56:32]
          | (2u << 30);                          // type = 2 ("image")
  i32x8 g1;
  g1[0] = (int)(1u << 16);       // workgroup_mask=0, data_size=1 (2 bytes)
  g1[1] = (int)(cols << 16);     // tensor_dim0[15:0] in bits 63:48
  g1[2] = (int)(rows << 16);     // tensor_dim0[31:16]=0 | tensor_dim1[15:0]
  g1[3] = (int)(cols << 16);     // tensor_dim1[31:16]=0 | tile_dim0
  g1[4] = (int)rows;             // tile_dim1 | tile_dim2=0
  g1[5] = (int)stride_elems;     // tensor_dim0_stride[31:0]
  g1[6] = 0;                     // stride hi | tensor_dim1_stride lo
  g1[7] = 0;
  const i32x4 z4 = {0, 0, 0, 0};            // groups 2/3 unused (2D tensor)
  const i32x8 z8 = {0, 0, 0, 0, 0, 0, 0, 0};
  __builtin_amdgcn_tensor_load_to_lds(g0, g1, z4, z4, z8, 0);
}

// ------------------------------------------------------- tiled WMMA bf16 GEMM
// C[m,n] = alpha * sum_k A[m,k] * Bop[k,n]  (+ bias[n])
//   TRANSB == 0 : Bop[k,n] = B[n*ldb + k]   (B row-major [N][K])
//   TRANSB == 1 : Bop[k,n] = B[k*ldb + n]   (B row-major [K][N])
// batchStrideB != 0 : B advances by batchStrideB per 512 rows of M.
// Block: 128 threads = 4 waves; 64x64 C tile; wave = 32x32 quadrant
// (2x2 WMMA fragments), K in steps of 32 through LDS. A staged by TDM.
template <int TRANSB, int OUTBF16>
__global__ __launch_bounds__(128)
void mha_gemm_wmma(const __bf16* __restrict__ A, const __bf16* __restrict__ B,
                   void* __restrict__ Cp, int Kdim, int lda, int ldb, int ldc,
                   long batchStrideB, float alpha,
                   const float* __restrict__ bias) {
  __shared__ __bf16 As[64 * 32];   // [m][k]
  __shared__ __bf16 Bs[64 * 32];   // TRANSB==0: [n][k] ; TRANSB==1: [k][n]

  const int tid  = threadIdx.x;
  const int lane = tid & 31;
  const int wave = tid >> 5;
  const int half = lane >> 4;
  const int l16  = lane & 15;
  const int wm   = wave >> 1;
  const int wn   = wave & 1;

  const long tileM = (long)blockIdx.y * 64;
  const int  tileN = blockIdx.x * 64;

  const __bf16* Bb = B;
  if (batchStrideB) Bb += (tileM / GRAPH) * batchStrideB;

  v8f acc[2][2];
  for (int a = 0; a < 2; ++a)
    for (int b = 0; b < 2; ++b)
      for (int r = 0; r < 8; ++r) acc[a][b][r] = 0.0f;

  const unsigned ldsAs = (unsigned)(uintptr_t)(void*)&As[0];
  const __bf16* Atile = A + tileM * (long)lda;

  const int kTiles = Kdim >> 5;
  for (int kt = 0; kt < kTiles; ++kt) {
    const int kbase = kt << 5;
    // ---- stage B tile (coalesced b128 global loads) ----
    if (!TRANSB) {
      const int r0 = tid >> 1;           // n row 0..63
      const int c0 = (tid & 1) * 16;     // k col 0/16
      const __bf16* bp = Bb + (long)(tileN + r0) * ldb + kbase + c0;
#pragma unroll
      for (int j = 0; j < 16; ++j) Bs[r0 * 32 + c0 + j] = bp[j];
      if (kt + 1 < kTiles) __builtin_prefetch(bp + 32, 0, 3);
    } else {
      const int r0 = tid >> 2;           // k row 0..31
      const int c0 = (tid & 3) * 16;     // n col 0/16/32/48
      const __bf16* bp = Bb + (long)(kbase + r0) * ldb + tileN + c0;
#pragma unroll
      for (int j = 0; j < 16; ++j) Bs[r0 * 64 + c0 + j] = bp[j];
      if (kt + 1 < kTiles) __builtin_prefetch(bp + 32 * ldb, 0, 3);
    }
    // ---- stage A tile via Tensor Data Mover (wave 0 issues once) ----
    if (wave == 0) {
      tdm_load_tile_bf16(ldsAs, Atile + kbase, 64, 32, (unsigned)lda);
      __builtin_amdgcn_s_wait_tensorcnt(0);
    }
    __syncthreads();

    // A fragment (16x32): lane m=l16; elems 0-7 -> K=half*8+i, 8-15 -> +16.
    v16bf afrag[2], bfrag[2];
#pragma unroll
    for (int fm = 0; fm < 2; ++fm) {
      const __bf16* rp = &As[(wm * 32 + fm * 16 + l16) * 32];
#pragma unroll
      for (int i = 0; i < 8; ++i) {
        afrag[fm][i]     = rp[half * 8 + i];
        afrag[fm][8 + i] = rp[16 + half * 8 + i];
      }
    }
    // B fragment (32x16): lane n=l16; elems 0-7 -> K=half*8+e, 8-15 -> +16.
#pragma unroll
    for (int fn = 0; fn < 2; ++fn) {
      const int nloc = wn * 32 + fn * 16 + l16;
#pragma unroll
      for (int e = 0; e < 16; ++e) {
        const int kk = half * 8 + e + (e >= 8 ? 8 : 0);
        bfrag[fn][e] = TRANSB ? Bs[kk * 64 + nloc]
                              : Bs[nloc * 32 + kk];
      }
    }
#pragma unroll
    for (int fm = 0; fm < 2; ++fm)
#pragma unroll
      for (int fn = 0; fn < 2; ++fn)
        acc[fm][fn] = __builtin_amdgcn_wmma_f32_16x16x32_bf16(
            false, afrag[fm], false, bfrag[fn], (short)0, acc[fm][fn],
            false, false);
    __syncthreads();
  }

  // C/D layout: VGPR r -> row (lane<16 ? r : r+8), col = l16.
#pragma unroll
  for (int fm = 0; fm < 2; ++fm) {
#pragma unroll
    for (int fn = 0; fn < 2; ++fn) {
      const int   gn = tileN + wn * 32 + fn * 16 + l16;
      const float bv = bias ? bias[gn] : 0.0f;
#pragma unroll
      for (int r = 0; r < 8; ++r) {
        const long gm = tileM + wm * 32 + fm * 16 + (half ? r + 8 : r);
        const float v = alpha * acc[fm][fn][r] + bv;
        if (OUTBF16) ((__bf16*)Cp)[gm * ldc + gn] = (__bf16)v;
        else         ((float*)Cp)[gm * ldc + gn]  = v;
      }
    }
  }
}

// ------------------------------------------------- masked softmax over keys
__global__ __launch_bounds__(256)
void mha_softmax_mask(const float* __restrict__ S,
                      const unsigned char* __restrict__ mask,
                      __bf16* __restrict__ P) {
  const int  wave = threadIdx.x >> 5;
  const int  lane = threadIdx.x & 31;
  const long row  = (long)blockIdx.x * 8 + wave;
  const int  b    = (int)(row / GRAPH);
  const float* srow = S + row * GRAPH;
  const unsigned char* mrow = mask + (long)b * GRAPH;

  float x[16];
  bool  msk[16];
  float mx = -INFINITY;
#pragma unroll
  for (int i = 0; i < 16; ++i) {
    const int k = lane + i * 32;
    msk[i] = mrow[k] != 0;
    x[i]   = msk[i] ? -INFINITY : srow[k];
    mx     = fmaxf(mx, x[i]);
  }
  for (int off = 16; off; off >>= 1) mx = fmaxf(mx, __shfl_xor(mx, off, 32));

  float e[16];
  float sum = 0.0f;
#pragma unroll
  for (int i = 0; i < 16; ++i) { e[i] = __expf(x[i] - mx); sum += e[i]; }
  for (int off = 16; off; off >>= 1) sum += __shfl_xor(sum, off, 32);

  const float inv = (sum > 0.0f) ? (1.0f / sum) : 0.0f;
  __bf16* prow = P + row * GRAPH;
#pragma unroll
  for (int i = 0; i < 16; ++i) {
    const int k = lane + i * 32;
    prow[k] = (__bf16)(msk[i] ? 0.0f : e[i] * inv);
  }
}

// -------------------------------------------------------------------- driver
extern "C" void kernel_launch(void* const* d_in, const int* in_sizes, int n_in,
                              void* d_out, int out_size, void* d_ws,
                              size_t ws_size, hipStream_t stream) {
  const float* data = (const float*)d_in[0];
  const float* Wq   = (const float*)d_in[1];
  const float* Wk   = (const float*)d_in[2];
  const float* Wv   = (const float*)d_in[3];
  const float* Wo   = (const float*)d_in[4];
  const float* Wob  = (const float*)d_in[5];
  const unsigned char* mask = (const unsigned char*)d_in[6];  // jax bool = 1B

  char* ws = (char*)d_ws;
  const size_t ROWEL = (size_t)ROWS * DIM;
  __bf16* dataB = (__bf16*)(ws);                  // 64MB; reused as P
  __bf16* Qb    = (__bf16*)(ws + ROWEL * 2);      // 64MB; reused as heads
  __bf16* Kb    = (__bf16*)(ws + ROWEL * 4);      // 64MB
  __bf16* Vb    = (__bf16*)(ws + ROWEL * 6);      // 64MB
  __bf16* WqB   = (__bf16*)(ws + ROWEL * 8);
  __bf16* WkB   = WqB + (size_t)DIM * DIM;
  __bf16* WvB   = WkB + (size_t)DIM * DIM;
  __bf16* WoB   = WvB + (size_t)DIM * DIM;
  float*  Sbuf  = (float*)(ws + ROWEL * 8 + (size_t)4 * DIM * DIM * 2);

  const int nData = ROWS * DIM;
  const int nW    = DIM * DIM;
  mha_cvt_bf16<<<(nData + 255) / 256, 256, 0, stream>>>(data, dataB, nData);
  mha_cvt_bf16<<<(nW + 255) / 256, 256, 0, stream>>>(Wq, WqB, nW);
  mha_cvt_bf16<<<(nW + 255) / 256, 256, 0, stream>>>(Wk, WkB, nW);
  mha_cvt_bf16<<<(nW + 255) / 256, 256, 0, stream>>>(Wv, WvB, nW);
  mha_cvt_bf16<<<(nW + 255) / 256, 256, 0, stream>>>(Wo, WoB, nW);

  const dim3 blk(128);
  const dim3 grd(DIM / 64, ROWS / 64);  // (8, 1024)
  const long bsB = (long)GRAPH * DIM;
  const float norm = 0.04419417382415922f;  // 1/sqrt(512)

  // Q/K/V = data @ W^T
  mha_gemm_wmma<0, 1><<<grd, blk, 0, stream>>>(dataB, WqB, Qb, DIM, DIM, DIM,
                                               DIM, 0, 1.0f, nullptr);
  mha_gemm_wmma<0, 1><<<grd, blk, 0, stream>>>(dataB, WkB, Kb, DIM, DIM, DIM,
                                               DIM, 0, 1.0f, nullptr);
  mha_gemm_wmma<0, 1><<<grd, blk, 0, stream>>>(dataB, WvB, Vb, DIM, DIM, DIM,
                                               DIM, 0, 1.0f, nullptr);
  // S = norm * Q @ K^T  (batched over graphs)
  mha_gemm_wmma<0, 0><<<grd, blk, 0, stream>>>(Qb, Kb, Sbuf, DIM, DIM, DIM,
                                               DIM, bsB, norm, nullptr);
  // P = softmax(mask(S)) -> bf16 (reuses data buffer)
  mha_softmax_mask<<<ROWS / 8, 256, 0, stream>>>(Sbuf, mask, dataB);
  // heads = P @ V  (batched, V is [keys][d] -> transposed staging)
  mha_gemm_wmma<1, 1><<<grd, blk, 0, stream>>>(dataB, Vb, Qb, DIM, DIM, DIM,
                                               DIM, bsB, 1.0f, nullptr);
  // out = heads @ Wo^T + bias  (fp32 to d_out)
  mha_gemm_wmma<0, 0><<<grd, blk, 0, stream>>>(Qb, WoB, d_out, DIM, DIM, DIM,
                                               DIM, 0, 1.0f, Wob);
}
